// AttentionBlock_60017872995084
// MI455X (gfx1250) — compile-verified, matrix operand order checked by reading, not run on hardware
//
#include <hip/hip_runtime.h>
#include <hip/hip_bf16.h>
#include <math.h>

typedef __attribute__((ext_vector_type(16))) _Float16 v16h;
typedef __attribute__((ext_vector_type(8)))  _Float16 v8h;
typedef __attribute__((ext_vector_type(8)))  float    v8f;
typedef __attribute__((ext_vector_type(4)))  float    v4f;

#define BSZ  8
#define CDIM 128
#define WDIM 2048

// ---------------------------------------------------------------------------
// Fragment loader: matrix stored row-major [rows][ld] f16, fragment rows = M/N,
// fragment K = contiguous columns. ISA 16-bit A/B layout: lane (m=l&15, h=l>>4)
// holds K = 8h..8h+7 in VGPRs 0-3 and K = 16+8h..16+8h+7 in VGPRs 4-7.
// ---------------------------------------------------------------------------
__device__ __forceinline__ v16h load_frag_f16(const _Float16* __restrict__ p) {
    v8h lo = *(const v8h*)p;         // K = 8h .. 8h+7
    v8h hi = *(const v8h*)(p + 16);  // K = 16+8h .. 16+8h+7
    v16h r;
#pragma unroll
    for (int t = 0; t < 8; ++t) { r[t] = lo[t]; r[t + 8] = hi[t]; }
    return r;
}

// ---------------------------------------------------------------------------
// Projection kernel: q/k/v = W*x + b via WMMA (grid = 256, 128 threads).
// Q,K stored [B][W][C] f16 (c contiguous), V stored [B][C][W] f16.
// ---------------------------------------------------------------------------
__global__ __launch_bounds__(128) void proj_kernel(
    const float* __restrict__ x,
    const float* __restrict__ Wq, const float* __restrict__ bq,
    const float* __restrict__ Wk, const float* __restrict__ bk,
    const float* __restrict__ Wv, const float* __restrict__ bv,
    _Float16* __restrict__ Q, _Float16* __restrict__ K, _Float16* __restrict__ V)
{
    const int LD = 136;                       // padded leading dim (halves)
    __shared__ _Float16 xT[64 * 136];         // x tile transposed: [w][c] f16

    const int b  = blockIdx.x >> 5;
    const int w0 = (blockIdx.x & 31) * 64;
    const int tid = threadIdx.x;

    for (int idx = tid; idx < 64 * 128; idx += 128) {
        int c = idx >> 6, w = idx & 63;
        float val = x[((size_t)b * CDIM + c) * WDIM + w0 + w];
        xT[w * LD + c] = (_Float16)val;
    }
    __syncthreads();

    const int lane = tid & 31, wave = tid >> 5;
    const int m = lane & 15, h = lane >> 4;
    const int wglob = w0 + wave * 16 + m;

    v16h bf[4];
#pragma unroll
    for (int kc = 0; kc < 4; ++kc)
        bf[kc] = load_frag_f16(&xT[(wave * 16 + m) * LD + kc * 32 + 8 * h]);

#pragma unroll
    for (int mat = 0; mat < 3; ++mat) {
        const float* Wm = (mat == 0) ? Wq : (mat == 1) ? Wk : Wv;
        const float* bm = (mat == 0) ? bq : (mat == 1) ? bk : bv;
#pragma unroll
        for (int ot = 0; ot < 8; ++ot) {
            v8f acc = {};
#pragma unroll
            for (int kc = 0; kc < 4; ++kc) {
                const float* p = Wm + (ot * 16 + m) * CDIM + kc * 32 + 8 * h;
                v4f a0 = *(const v4f*)p;
                v4f a1 = *(const v4f*)(p + 4);
                v4f c0 = *(const v4f*)(p + 16);
                v4f c1 = *(const v4f*)(p + 20);
                v16h af;
#pragma unroll
                for (int t = 0; t < 4; ++t) {
                    af[t]      = (_Float16)a0[t];
                    af[4 + t]  = (_Float16)a1[t];
                    af[8 + t]  = (_Float16)c0[t];
                    af[12 + t] = (_Float16)c1[t];
                }
                acc = __builtin_amdgcn_wmma_f32_16x16x32_f16(
                          false, af, false, bf[kc], (short)0, acc, false, false);
            }
            const int ob = ot * 16 + 8 * h;
#pragma unroll
            for (int r = 0; r < 8; ++r) acc[r] += bm[ob + r];

            if (mat < 2) {
                _Float16* dst = (mat == 0) ? Q : K;
                v8h pk;
#pragma unroll
                for (int r = 0; r < 8; ++r) pk[r] = (_Float16)acc[r];
                *(v8h*)&dst[((size_t)b * WDIM + wglob) * CDIM + ob] = pk;
            } else {
#pragma unroll
                for (int r = 0; r < 8; ++r)
                    V[((size_t)b * CDIM + ob + r) * WDIM + wglob] = (_Float16)acc[r];
            }
        }
    }
}

// ---------------------------------------------------------------------------
// Flash-attention kernel: 4 waves/block, one 16-query tile per wave (same batch),
// K/V tiles double-buffered in LDS via async global->LDS DMA (ASYNCcnt).
// ---------------------------------------------------------------------------
__global__ __launch_bounds__(128) void flash_kernel(
    const _Float16* __restrict__ Q, const _Float16* __restrict__ Kt,
    const _Float16* __restrict__ V, const float* __restrict__ x,
    const float* __restrict__ gamma_p, float* __restrict__ out)
{
    __shared__ _Float16 kbuf[2][32 * 128];   // [j-local][c]   8 KB each
    __shared__ _Float16 vbuf[2][128 * 32];   // [c][j-local]   8 KB each

    const int tid  = threadIdx.x;
    const int wave = tid >> 5, lane = tid & 31;
    const int m = lane & 15, h = lane >> 4;
    const int tile = blockIdx.x * 4 + wave;   // 128 tiles/batch -> whole block same b
    const int b  = tile >> 7;
    const int i0 = (tile & 127) * 16;

    const _Float16* kb = Kt + (size_t)b * WDIM * CDIM;
    const _Float16* vb = V  + (size_t)b * CDIM * WDIM;

    // Issue async DMA of one 32-key block of K (contiguous 8KB) and V (128 x 64B)
    auto issue_async = [&](int buf, int j0) {
        const char* kg = (const char*)(kb + (size_t)j0 * CDIM);
        unsigned    kl = (unsigned)(size_t)&kbuf[buf][0];
#pragma unroll
        for (int r = 0; r < 4; ++r) {
            unsigned    loff = kl + tid * 16 + r * 2048;
            const char* g    = kg + tid * 16 + r * 2048;
            asm volatile("global_load_async_to_lds_b128 %0, %1, off"
                         :: "v"(loff), "v"(g) : "memory");
        }
        const char* vg = (const char*)(vb + (size_t)tid * WDIM + j0);
        unsigned    vl = (unsigned)(size_t)&vbuf[buf][tid * 32];
#pragma unroll
        for (int r = 0; r < 4; ++r) {
            unsigned    loff = vl + r * 16;
            const char* g    = vg + r * 16;
            asm volatile("global_load_async_to_lds_b128 %0, %1, off"
                         :: "v"(loff), "v"(g) : "memory");
        }
    };

    // Q B-fragments (column i = i0+m), hoisted for the whole key loop
    const _Float16* qrow = Q + ((size_t)b * WDIM + i0 + m) * CDIM;
    v16h qf[4];
#pragma unroll
    for (int kc = 0; kc < 4; ++kc) qf[kc] = load_frag_f16(qrow + kc * 32 + 8 * h);

    v8f o_acc[8];
#pragma unroll
    for (int ct = 0; ct < 8; ++ct) o_acc[ct] = (v8f){};
    float m_run = -INFINITY;
    float l_run = 0.0f;

    issue_async(0, 0);                        // prologue DMA for block 0

    for (int t = 0; t < WDIM / 32; ++t) {
        asm volatile("s_wait_asynccnt 0x0" ::: "memory");
        __syncthreads();                      // buf[t&1] visible to all waves
        if (t + 1 < WDIM / 32)
            issue_async((t + 1) & 1, (t + 1) * 32);   // overlap next DMA with compute

        const _Float16* kt = &kbuf[t & 1][0];
        const _Float16* vt = &vbuf[t & 1][0];

        // ---- GEMM1: E^T tiles for key rows 0..15 (e0) and 16..31 (e1) of this block
        v8f e0 = {}, e1 = {};
#pragma unroll
        for (int kc = 0; kc < 4; ++kc) {
            v16h ka0 = load_frag_f16(kt + m * 128        + kc * 32 + 8 * h);
            v16h ka1 = load_frag_f16(kt + (16 + m) * 128 + kc * 32 + 8 * h);
            e0 = __builtin_amdgcn_wmma_f32_16x16x32_f16(false, ka0, false, qf[kc], (short)0, e0, false, false);
            e1 = __builtin_amdgcn_wmma_f32_16x16x32_f16(false, ka1, false, qf[kc], (short)0, e1, false, false);
        }

        // ---- online softmax over this block's 32 keys (per query i = lane&15)
        float bmax = fmaxf(e0[0], e1[0]);
#pragma unroll
        for (int r = 1; r < 8; ++r) bmax = fmaxf(bmax, fmaxf(e0[r], e1[r]));
        bmax = fmaxf(bmax, __shfl_xor(bmax, 16, 32));
        const float m_new = fmaxf(m_run, bmax);
        const float scale = __expf(m_run - m_new);

        float psum = 0.0f;
        v16h pf;                              // P^T B-fragment, built in-lane
#pragma unroll
        for (int r = 0; r < 8; ++r) {
            float p0 = __expf(e0[r] - m_new); psum += p0; pf[r]     = (_Float16)p0;
            float p1 = __expf(e1[r] - m_new); psum += p1; pf[8 + r] = (_Float16)p1;
        }
        psum += __shfl_xor(psum, 16, 32);
        l_run = l_run * scale + psum;
        m_run = m_new;

        // ---- rescale O and accumulate GEMM2: O^T[c,i] += V[c,j] * P^T[j,i]
#pragma unroll
        for (int ct = 0; ct < 8; ++ct) {
#pragma unroll
            for (int r = 0; r < 8; ++r) o_acc[ct][r] *= scale;
            v16h vf = load_frag_f16(vt + (ct * 16 + m) * 32 + 8 * h);
            o_acc[ct] = __builtin_amdgcn_wmma_f32_16x16x32_f16(
                            false, vf, false, pf, (short)0, o_acc[ct], false, false);
        }
    }

    // ---- epilogue: out = gamma * (O / l) + x
    const float inv_l = 1.0f / l_run;
    const float g = gamma_p[0];
#pragma unroll
    for (int ct = 0; ct < 8; ++ct) {
#pragma unroll
        for (int r = 0; r < 8; ++r) {
            size_t idx = ((size_t)b * CDIM + ct * 16 + 8 * h + r) * WDIM + i0 + m;
            out[idx] = g * (o_acc[ct][r] * inv_l) + x[idx];
        }
    }
}

// ---------------------------------------------------------------------------
extern "C" void kernel_launch(void* const* d_in, const int* in_sizes, int n_in,
                              void* d_out, int out_size, void* d_ws, size_t ws_size,
                              hipStream_t stream) {
    const float* x     = (const float*)d_in[0];
    const float* Wq    = (const float*)d_in[1];
    const float* bq    = (const float*)d_in[2];
    const float* Wk    = (const float*)d_in[3];
    const float* bk    = (const float*)d_in[4];
    const float* Wv    = (const float*)d_in[5];
    const float* bv    = (const float*)d_in[6];
    const float* gamma = (const float*)d_in[7];
    float* out = (float*)d_out;

    const size_t elems = (size_t)BSZ * WDIM * CDIM;   // 2M halves each
    _Float16* Qh = (_Float16*)d_ws;
    _Float16* Kh = Qh + elems;
    _Float16* Vh = Kh + elems;                        // 12 MB total f16 workspace

    proj_kernel<<<BSZ * (WDIM / 64), 128, 0, stream>>>(x, Wq, bq, Wk, bk, Wv, bv,
                                                       Qh, Kh, Vh);
    flash_kernel<<<(BSZ * (WDIM / 16)) / 4, 128, 0, stream>>>(Qh, Kh, Vh, x, gamma, out);
}
